// kernel_layer_30408368455724
// MI455X (gfx1250) — matrix operand order repaired and checked by reading, not compiled, not run on hardware
//
#include <hip/hip_runtime.h>
#include <cstddef>

#define GAMMA 0.002f
#define TILE 128      // output tile (M and N) per block
#define KD 512        // feature dim (fixed by the reference)
#define KC 32         // K-chunk per WMMA step
#define RS 40         // LDS row stride in bf16 elems (80B: 16B-aligned, bank stride 20)

typedef __attribute__((ext_vector_type(16))) __bf16 v16bf;
typedef __attribute__((ext_vector_type(8)))  __bf16 bf16x8;
typedef __attribute__((ext_vector_type(8)))  float  v8f;

__device__ __forceinline__ v16bf combine8(bf16x8 a, bf16x8 b) {
    v16bf r;
#pragma unroll
    for (int i = 0; i < 8; ++i) { r[i] = a[i]; r[i + 8] = b[i]; }
    return r;
}

__global__ __launch_bounds__(256) void rbf_wmma_kernel(
    const float* __restrict__ X,   // [B, 512]
    const float* __restrict__ S,   // [N, 512]
    float* __restrict__ O,         // [B, N]
    int Ncols)
{
    __shared__ __attribute__((aligned(16))) __bf16 Ahi[TILE * RS];
    __shared__ __attribute__((aligned(16))) __bf16 Alo[TILE * RS];
    __shared__ __attribute__((aligned(16))) __bf16 Bhi[TILE * RS];
    __shared__ __attribute__((aligned(16))) __bf16 Blo[TILE * RS];
    __shared__ float nparA[256], nparB[256];
    __shared__ float x2s[TILE], s2s[TILE];

    const int t    = threadIdx.x;
    const int lane = t & 31;
    const int wave = t >> 5;
    const int wm   = wave >> 2;          // 0..1 : 64-row half of tile
    const int wn   = wave & 3;           // 0..3 : 32-col quarter of tile

    const int gm = blockIdx.y * TILE;
    const int gn = blockIdx.x * TILE;

    // cooperative load mapping: thread -> (row 0..127, 16-float half of 32-chunk)
    const int lrow  = t >> 1;
    const int lkoff = (t & 1) * 16;
    const float* __restrict__ xrow = X + (size_t)(gm + lrow) * KD + lkoff;
    const float* __restrict__ srow = S + (size_t)(gn + lrow) * KD + lkoff;

    float anorm = 0.f, bnorm = 0.f;

    v8f acc[4][2] = {};                  // 4 m-subtiles x 2 n-subtiles of 16x16

    // A fragment addressing (16-bit A 16x32 ISA layout)
    const int fr  = lane & 15;           // row within 16
    const int fka = (lane >> 4) * 8;     // K base: 0 or 8 (pairs with +16)
    // B fragment addressing (16-bit B 32x16 ISA layout)
    const int fkb = (lane >> 4) * 8;     // K base: 0 or 8 (pairs with +16)

    for (int kc = 0; kc < KD; kc += KC) {
        __syncthreads();                 // protect LDS reuse across iterations

        const float4* xp = (const float4*)(xrow + kc);
        const float4* sp = (const float4*)(srow + kc);
        float4 xv[4], sv4[4];
#pragma unroll
        for (int v = 0; v < 4; ++v) { xv[v] = xp[v]; sv4[v] = sp[v]; }

        bf16x8 hA[2], lA[2], hB[2], lB[2];
#pragma unroll
        for (int v = 0; v < 4; ++v) {
#pragma unroll
            for (int e = 0; e < 4; ++e) {
                const int idx = v * 4 + e;
                float fx = ((const float*)&xv[v])[e];
                float fs = ((const float*)&sv4[v])[e];
                anorm += fx * fx;
                bnorm += fs * fs;
                __bf16 hx = (__bf16)fx;
                __bf16 hs = (__bf16)fs;
                hA[idx >> 3][idx & 7] = hx;
                lA[idx >> 3][idx & 7] = (__bf16)(fx - (float)hx);
                hB[idx >> 3][idx & 7] = hs;
                lB[idx >> 3][idx & 7] = (__bf16)(fs - (float)hs);
            }
        }

        __bf16* pa = &Ahi[lrow * RS + lkoff];
        __bf16* qa = &Alo[lrow * RS + lkoff];
        __bf16* pb = &Bhi[lrow * RS + lkoff];
        __bf16* qb = &Blo[lrow * RS + lkoff];
        *(bf16x8*)(pa) = hA[0]; *(bf16x8*)(pa + 8) = hA[1];
        *(bf16x8*)(qa) = lA[0]; *(bf16x8*)(qa + 8) = lA[1];
        *(bf16x8*)(pb) = hB[0]; *(bf16x8*)(pb + 8) = hB[1];
        *(bf16x8*)(qb) = lB[0]; *(bf16x8*)(qb + 8) = lB[1];

        __syncthreads();

        // A fragments: lane holds row fr, K {fka..fka+7} and {fka+16..fka+23}
        v16bf aH[4], aL[4];
#pragma unroll
        for (int mi = 0; mi < 4; ++mi) {
            const int r = wm * 64 + mi * 16 + fr;
            const __bf16* ph = &Ahi[r * RS + fka];
            const __bf16* pl = &Alo[r * RS + fka];
            aH[mi] = combine8(*(const bf16x8*)ph, *(const bf16x8*)(ph + 16));
            aL[mi] = combine8(*(const bf16x8*)pl, *(const bf16x8*)(pl + 16));
        }
        // B fragments: lane holds col fr, K {fkb..fkb+7} and {fkb+16..fkb+23}
        v16bf bH[2], bL[2];
#pragma unroll
        for (int ni = 0; ni < 2; ++ni) {
            const int r = wn * 32 + ni * 16 + fr;
            const __bf16* ph = &Bhi[r * RS + fkb];
            const __bf16* pl = &Blo[r * RS + fkb];
            bH[ni] = combine8(*(const bf16x8*)ph, *(const bf16x8*)(ph + 16));
            bL[ni] = combine8(*(const bf16x8*)pl, *(const bf16x8*)(pl + 16));
        }

        // split-bf16 GEMM: cross ~= hi*hi + hi*lo + lo*hi
#pragma unroll
        for (int mi = 0; mi < 4; ++mi) {
#pragma unroll
            for (int ni = 0; ni < 2; ++ni) {
                acc[mi][ni] = __builtin_amdgcn_wmma_f32_16x16x32_bf16(
                    false, aH[mi], false, bH[ni], (short)0, acc[mi][ni], false, false);
                acc[mi][ni] = __builtin_amdgcn_wmma_f32_16x16x32_bf16(
                    false, aH[mi], false, bL[ni], (short)0, acc[mi][ni], false, false);
                acc[mi][ni] = __builtin_amdgcn_wmma_f32_16x16x32_bf16(
                    false, aL[mi], false, bH[ni], (short)0, acc[mi][ni], false, false);
            }
        }
    }

    // reduce half-row norm partials -> per-row ||x||^2, ||sv||^2 for this tile
    nparA[t] = anorm;
    nparB[t] = bnorm;
    __syncthreads();
    if (t < TILE) {
        x2s[t] = nparA[2 * t] + nparA[2 * t + 1];
        s2s[t] = nparB[2 * t] + nparB[2 * t + 1];
    }
    __syncthreads();

    // epilogue: d2 = x2 + s2 - 2*cross ; out = exp(-gamma*d2)
    const int rhalf = 8 * (lane >> 4);
    const int ccol  = lane & 15;
#pragma unroll
    for (int mi = 0; mi < 4; ++mi) {
#pragma unroll
        for (int ni = 0; ni < 2; ++ni) {
            const int cl = wn * 32 + ni * 16 + ccol;
            const float s2v = s2s[cl];
#pragma unroll
            for (int i = 0; i < 8; ++i) {
                const int rl = wm * 64 + mi * 16 + i + rhalf;
                const float d2 = x2s[rl] + s2v - 2.0f * acc[mi][ni][i];
                O[(size_t)(gm + rl) * Ncols + (gn + cl)] = __expf(-GAMMA * d2);
            }
        }
    }
}

extern "C" void kernel_launch(void* const* d_in, const int* in_sizes, int n_in,
                              void* d_out, int out_size, void* d_ws, size_t ws_size,
                              hipStream_t stream) {
    const float* X = (const float*)d_in[0];   // x  [B, 512] f32
    const float* S = (const float*)d_in[1];   // sv [N, 512] f32
    float* O = (float*)d_out;                 // K  [B, N]  f32

    const int D = KD;
    const int B = in_sizes[0] / D;            // 4096
    const int N = in_sizes[1] / D;            // 8192

    dim3 grid(N / TILE, B / TILE);            // (64, 32)
    rbf_wmma_kernel<<<grid, 256, 0, stream>>>(X, S, O, N);
}
